// SelfAttention_64579128262907
// MI455X (gfx1250) — matrix operand order repaired and checked by reading, not compiled, not run on hardware
//
#include <hip/hip_runtime.h>
#include <hip/hip_bf16.h>

// ---------------------------------------------------------------------------
// Self-attention (B=8, N=2048, D=768, softmax over axis=-2) for gfx1250.
// All three GEMMs on v_wmma_f32_16x16x32_bf16; LDS-staged operands with
// bank-conflict padding; async global->LDS staging; prefetch on streams.
// ---------------------------------------------------------------------------

typedef __bf16 bf16_t;
typedef bf16_t v16bf __attribute__((ext_vector_type(16)));
typedef bf16_t v8bf  __attribute__((ext_vector_type(8)));
typedef float  v8f   __attribute__((ext_vector_type(8)));

#define BATCH   8
#define SEQ     2048
#define DIM     768
#define E3      2304                  // 3*DIM
#define SCALE   0.0360843918243516f   // 1/sqrt(768)

#define LDK     776                   // padded LDS row stride for 768-elem rows
#define LDP     136                   // padded LDS row stride for 128-elem rows

static __device__ __forceinline__ bf16_t cvt_bf16(float f) {
    unsigned u = __builtin_bit_cast(unsigned, f);
    u += 0x7FFFu + ((u >> 16) & 1u);              // round-to-nearest-even
    unsigned short h = (unsigned short)(u >> 16);
    return __builtin_bit_cast(bf16_t, h);
}

// ---- WMMA wrapper ----------------------------------------------------------
static __device__ __forceinline__ v8f wmma_bf16(v16bf a, v16bf b, v8f c) {
    return __builtin_amdgcn_wmma_f32_16x16x32_bf16(
        false, a, false, b, (short)0, c, false, false);
}

// ---- Fragment loaders (all via 16B v8bf loads -> b128) ---------------------
// A (16x32 bf16, row-major, stride ld):
//   lanes 0-15 : M=lane,    elems 0-15 = K k0..+15
//   lanes16-31 : M=lane-16, elems 0-15 = K k0+16..+31
static __device__ __forceinline__ v16bf load_a_frag(const bf16_t* src, int ld,
                                                    int row0, int k0) {
    int lane = threadIdx.x & 31;
    int m    = lane & 15;
    int hi   = lane >> 4;
    const bf16_t* p = src + (size_t)(row0 + m) * ld + k0 + hi * 16;
    v8bf lo  = *(const v8bf*)p;
    v8bf hi8 = *(const v8bf*)(p + 8);
    return __builtin_shufflevector(lo, hi8, 0,1,2,3,4,5,6,7,8,9,10,11,12,13,14,15);
}

// B (32x16 bf16): element (k,n) at src[n*ld + k]; col = lane&15,
// K = k0 + (lane>=16 ? 16 : 0) + 0..15 (two 16B chunks).
static __device__ __forceinline__ v16bf load_b_frag(const bf16_t* src, int ld,
                                                    int n0, int k0) {
    int lane = threadIdx.x & 31;
    int n    = lane & 15;
    int hi   = lane >> 4;
    const bf16_t* p = src + (size_t)(n0 + n) * ld + k0 + hi * 16;
    v8bf lo  = *(const v8bf*)p;
    v8bf hi8 = *(const v8bf*)(p + 8);
    return __builtin_shufflevector(lo, hi8, 0,1,2,3,4,5,6,7,8,9,10,11,12,13,14,15);
}

// ---------------------------------------------------------------------------
// Kernel 0: E fp32 -> bf16, vectorized 8 elems/thread.
// ---------------------------------------------------------------------------
__global__ void cvt_e_kernel(const float* __restrict__ E, bf16_t* __restrict__ Eb) {
    int i = (blockIdx.x * blockDim.x + threadIdx.x) * 8;
    if (i >= BATCH * SEQ * DIM) return;
    float4 f0 = *(const float4*)(E + i);
    float4 f1 = *(const float4*)(E + i + 4);
    v8bf o;
    o[0] = cvt_bf16(f0.x); o[1] = cvt_bf16(f0.y);
    o[2] = cvt_bf16(f0.z); o[3] = cvt_bf16(f0.w);
    o[4] = cvt_bf16(f1.x); o[5] = cvt_bf16(f1.y);
    o[6] = cvt_bf16(f1.z); o[7] = cvt_bf16(f1.w);
    *(v8bf*)(Eb + i) = o;
}

// ---------------------------------------------------------------------------
// Kernel 1: W (768x2304 fp32) -> Wt (2304x768 bf16), transposed.
// ---------------------------------------------------------------------------
__global__ void transpose_w_kernel(const float* __restrict__ W, bf16_t* __restrict__ Wt) {
    int idx = blockIdx.x * blockDim.x + threadIdx.x;
    if (idx >= DIM * E3) return;
    int d = idx / E3;
    int n = idx % E3;
    Wt[(size_t)n * DIM + d] = cvt_bf16(W[(size_t)d * E3 + n]);
}

// ---------------------------------------------------------------------------
// Kernel 2: QKV = E @ W. Each wave: 16 rows x 64 cols (A reused over 4 WMMAs).
// Q,K row-major bf16; V transposed (d x q). grid = (36, 128), block = 256.
// ---------------------------------------------------------------------------
__global__ void qkv_proj_kernel(const bf16_t* __restrict__ Eb, const bf16_t* __restrict__ Wt,
                                bf16_t* __restrict__ Q, bf16_t* __restrict__ Km,
                                bf16_t* __restrict__ Vt) {
    int w     = threadIdx.x >> 5;
    int lane  = threadIdx.x & 31;
    int rBase = blockIdx.y * 128 + w * 16;    // global row (b*2048 + q)
    int nGrp  = blockIdx.x * 64;              // output column group

    v8f acc[4] = {};
#pragma unroll 2
    for (int k0 = 0; k0 < DIM; k0 += 32) {
        v16bf a = load_a_frag(Eb, DIM, rBase, k0);
        __builtin_prefetch(Wt + (size_t)(nGrp + (lane & 15)) * DIM + k0 + 64, 0, 1);
#pragma unroll
        for (int t = 0; t < 4; ++t) {
            v16bf b = load_b_frag(Wt, DIM, nGrp + t * 16, k0);
            acc[t] = wmma_bf16(a, b, acc[t]);
        }
    }

    int rowOff = rBase + ((lane >> 4) << 3);  // 8 rows per lane, C-layout
#pragma unroll
    for (int t = 0; t < 4; ++t) {
        int col = nGrp + t * 16 + (lane & 15);
        if (col < DIM) {                      // Q
#pragma unroll
            for (int r = 0; r < 8; ++r)
                Q[(size_t)(rowOff + r) * DIM + col] = cvt_bf16(acc[t][r]);
        } else if (col < 2 * DIM) {           // K
#pragma unroll
            for (int r = 0; r < 8; ++r)
                Km[(size_t)(rowOff + r) * DIM + (col - DIM)] = cvt_bf16(acc[t][r]);
        } else {                              // V, transposed store (d x q)
            int d = col - 2 * DIM;
            int b = rowOff >> 11;
            int q = rowOff & 2047;
            v8bf pv;
#pragma unroll
            for (int r = 0; r < 8; ++r) pv[r] = cvt_bf16(acc[t][r]);
            *(v8bf*)(Vt + (size_t)b * DIM * SEQ + (size_t)d * SEQ + q) = pv;
        }
    }
}

// ---------------------------------------------------------------------------
// Kernel 3: per-key softmax stats (softmax over the q axis). Each wave owns
// 16 keys; K-tile staged in padded LDS (reused by 128 q-iterations).
// grid = (64, 8), block = 64 (2 waves).
// ---------------------------------------------------------------------------
__global__ void colstats_kernel(const bf16_t* __restrict__ Q, const bf16_t* __restrict__ Km,
                                float* __restrict__ Ms, float* __restrict__ Ls) {
    __shared__ __align__(16) bf16_t sK[2 * 16 * LDK];    // 48.5 KB

    int w       = threadIdx.x >> 5;
    int lane    = threadIdx.x & 31;
    int b       = blockIdx.y;
    int keyBase = (blockIdx.x * 2 + w) * 16;

    const bf16_t* Qb = Q  + (size_t)b * SEQ * DIM;
    const bf16_t* Kb = Km + (size_t)b * SEQ * DIM;
    bf16_t* sKw = sK + w * 16 * LDK;

    // stage this wave's 16 K rows into padded LDS
#pragma unroll
    for (int r = 0; r < 16; ++r)
        for (int c = lane * 8; c < DIM; c += 32 * 8)
            *(v8bf*)(sKw + r * LDK + c) = *(const v8bf*)(Kb + (size_t)(keyBase + r) * DIM + c);
    __syncthreads();

    float m_run = -3.0e38f;
    float l_run = 0.0f;

    for (int qb = 0; qb < SEQ; qb += 16) {
        v8f acc = {};
        __builtin_prefetch(Qb + (size_t)(qb + 16 + (lane & 15)) * DIM, 0, 1);
#pragma unroll 4
        for (int k0 = 0; k0 < DIM; k0 += 32) {
            v16bf a  = load_a_frag(Qb, DIM, qb, k0);
            v16bf bb = load_b_frag(sKw, LDK, 0, k0);
            acc = wmma_bf16(a, bb, acc);
        }
        float mx = m_run;
#pragma unroll
        for (int r = 0; r < 8; ++r) mx = fmaxf(mx, acc[r] * SCALE);
        float resc = __expf(m_run - mx);
        float s = 0.0f;
#pragma unroll
        for (int r = 0; r < 8; ++r) s += __expf(acc[r] * SCALE - mx);
        l_run = l_run * resc + s;
        m_run = mx;
    }

    float m_o = __shfl_xor(m_run, 16);
    float l_o = __shfl_xor(l_run, 16);
    float m_f = fmaxf(m_run, m_o);
    float l_f = l_run * __expf(m_run - m_f) + l_o * __expf(m_o - m_f);
    if (lane < 16) {
        Ms[(size_t)b * SEQ + keyBase + lane] = m_f;
        Ls[(size_t)b * SEQ + keyBase + lane] = l_f;
    }
}

// ---------------------------------------------------------------------------
// Kernel 4: out[q,d] = sum_k exp(s[q,k]-m[k])/l[k] * V[k,d].
// Block (4 waves) owns 16 q rows. Q-tile async-staged into LDS once; per
// 128-key block each wave builds a 16x32 P slice (WMMA + softmax -> LDS),
// then accumulates its 192 d-cols via 12 WMMAs per 32-key step. grid=(128,8).
// ---------------------------------------------------------------------------
__global__ void attn_out_kernel(const bf16_t* __restrict__ Q, const bf16_t* __restrict__ Km,
                                const bf16_t* __restrict__ Vt, const float* __restrict__ Ms,
                                const float* __restrict__ Ls, float* __restrict__ Out) {
    __shared__ __align__(16) bf16_t sQ[16 * LDK];   // 24.25 KB, LDS offset 0
    __shared__ __align__(16) bf16_t sP[16 * LDP];   // 4.25 KB

    int w     = threadIdx.x >> 5;
    int lane  = threadIdx.x & 31;
    int b     = blockIdx.y;
    int qBase = blockIdx.x * 16;
    int dBase = w * 192;

    const bf16_t* Qb = Q  + (size_t)b * SEQ * DIM;
    const bf16_t* Kb = Km + (size_t)b * SEQ * DIM;
    const bf16_t* Vb = Vt + (size_t)b * DIM * SEQ;
    const float* mrow = Ms + (size_t)b * SEQ;
    const float* lrow = Ls + (size_t)b * SEQ;

    // ---- async-stage the 16x768 Q tile into padded LDS (ASYNCcnt path) ----
    // 16 rows x 96 8-elem chunks = 1536 chunks, 12 per thread.
    for (int chunk = threadIdx.x; chunk < 16 * 96; chunk += 128) {
        int r = chunk / 96;
        int c = (chunk % 96) * 8;
        unsigned lds_off = (unsigned)((r * LDK + c) * sizeof(bf16_t));  // sQ at LDS base
        const bf16_t* gp = Qb + (size_t)(qBase + r) * DIM + c;
        asm volatile("global_load_async_to_lds_b128 %0, %1, off"
                     :: "v"(lds_off), "v"(gp) : "memory");
    }
    asm volatile("s_wait_asynccnt 0x0" ::: "memory");
    __syncthreads();

    v8f acc[12] = {};

    for (int kb = 0; kb < SEQ; kb += 128) {
        // ---- phase A: this wave produces P for keys [kb + w*32, +32) ----
#pragma unroll
        for (int t = 0; t < 2; ++t) {
            int keyB = kb + w * 32 + t * 16;
            v8f s = {};
#pragma unroll 4
            for (int k0 = 0; k0 < DIM; k0 += 32) {
                v16bf a  = load_a_frag(sQ, LDK, 0, k0);
                v16bf bb = load_b_frag(Kb, DIM, keyB, k0);
                s = wmma_bf16(a, bb, s);
            }
            int   key   = keyB + (lane & 15);
            float mv    = mrow[key];
            float inv_l = 1.0f / lrow[key];
            int   qOff  = (lane >> 4) << 3;
            int   cLoc  = w * 32 + t * 16 + (lane & 15);
#pragma unroll
            for (int r = 0; r < 8; ++r) {
                float p = __expf(s[r] * SCALE - mv) * inv_l;
                sP[(qOff + r) * LDP + cLoc] = cvt_bf16(p);
            }
        }
        __syncthreads();

        // ---- phase B: all waves consume P against V^T ----
        if (kb + 128 < SEQ)
            __builtin_prefetch(Kb + (size_t)(kb + 128 + w * 32 + (lane & 15)) * DIM, 0, 1);
#pragma unroll
        for (int sub = 0; sub < 4; ++sub) {
            v16bf a = load_a_frag(sP, LDP, 0, sub * 32);
#pragma unroll
            for (int dt = 0; dt < 12; ++dt) {
                v16bf bb = load_b_frag(Vb, SEQ, dBase + dt * 16, kb + sub * 32);
                acc[dt] = wmma_bf16(a, bb, acc[dt]);
            }
        }
        __syncthreads();
    }

    // ---- store 16q x 192d fp32 ----
    float* Ob  = Out + (size_t)b * SEQ * DIM;
    int    qOf = qBase + ((lane >> 4) << 3);
#pragma unroll
    for (int dt = 0; dt < 12; ++dt) {
        int col = dBase + dt * 16 + (lane & 15);
#pragma unroll
        for (int r = 0; r < 8; ++r)
            Ob[(size_t)(qOf + r) * DIM + col] = acc[dt][r];
    }
}

// ---------------------------------------------------------------------------
extern "C" void kernel_launch(void* const* d_in, const int* in_sizes, int n_in,
                              void* d_out, int out_size, void* d_ws, size_t ws_size,
                              hipStream_t stream) {
    (void)in_sizes; (void)n_in; (void)out_size; (void)ws_size;

    const float* E = (const float*)d_in[0];   // (8, 2048, 768) fp32
    const float* W = (const float*)d_in[1];   // (1, 768, 2304) fp32
    float* Out     = (float*)d_out;           // (8, 2048, 768) fp32

    char* ws = (char*)d_ws;
    const size_t szWt  = (size_t)E3 * DIM * sizeof(bf16_t);           // 3.54 MB
    const size_t szMat = (size_t)BATCH * SEQ * DIM * sizeof(bf16_t);  // 25.2 MB
    const size_t szVec = (size_t)BATCH * SEQ * sizeof(float);         // 64 KB

    bf16_t* Wt = (bf16_t*)(ws);
    bf16_t* Eb = (bf16_t*)(ws + szWt);
    bf16_t* Q  = (bf16_t*)(ws + szWt + szMat);
    bf16_t* Km = (bf16_t*)(ws + szWt + 2 * szMat);
    bf16_t* Vt = (bf16_t*)(ws + szWt + 3 * szMat);
    float*  Ms = (float*)(ws + szWt + 4 * szMat);
    float*  Ls = (float*)(ws + szWt + 4 * szMat + szVec);

    cvt_e_kernel<<<dim3((BATCH * SEQ * DIM / 8 + 255) / 256), dim3(256), 0, stream>>>(E, Eb);
    transpose_w_kernel<<<dim3((DIM * E3 + 255) / 256), dim3(256), 0, stream>>>(W, Wt);
    qkv_proj_kernel<<<dim3(E3 / 64, (BATCH * SEQ) / 128), dim3(256), 0, stream>>>(Eb, Wt, Q, Km, Vt);
    colstats_kernel<<<dim3(SEQ / 32, BATCH), dim3(64), 0, stream>>>(Q, Km, Ms, Ls);
    attn_out_kernel<<<dim3(SEQ / 16, BATCH), dim3(128), 0, stream>>>(Q, Km, Vt, Ms, Ls, Out);
}
